// LigerLMHeadGRPO_5153960755980
// MI455X (gfx1250) — matrix operand order repaired and checked by reading, not compiled, not run on hardware
//
#include <hip/hip_runtime.h>
#include <hip/hip_bf16.h>
#include <math.h>

// ---------------- problem constants ----------------
#define H_DIM 2048
#define V_DIM 32000
#define B_DIM 8
#define T_DIM 256
#define M_DIM (B_DIM * T_DIM)      // 2048 flattened rows
#define G_GEN 4
#define BETA_C 0.1f
#define EPS_C 1e-4f

// ---------------- tiling ----------------
#define BM 64                       // rows per workgroup (4 row-tiles)
#define BN 256                      // cols per inner iteration (8 waves x 32)
#define KS 32                       // K step = bf16 WMMA K
#define NK (H_DIM / KS)             // 64 K-steps
#define NITER 5                     // col iterations per vocab chunk
#define VCHUNK (BN * NITER)         // 1280
#define NCHUNK (V_DIM / VCHUNK)     // 25
#define NROWB (M_DIM / BM)          // 32
#define PSTRIDE 8                   // floats per (row, chunk) partial record
#define NEG_HUGE (-3.0e38f)

#define AS1 __attribute__((address_space(1)))
#define AS3 __attribute__((address_space(3)))

#if __has_builtin(__builtin_amdgcn_global_load_async_to_lds_b128) && \
    __has_builtin(__builtin_amdgcn_s_wait_asynccnt)
#define HAVE_ASYNC_LDS 1
#else
#define HAVE_ASYNC_LDS 0
#endif

typedef __attribute__((ext_vector_type(16))) __bf16 v16bf;
typedef __attribute__((ext_vector_type(8)))  __bf16 v8bf;
typedef __attribute__((ext_vector_type(8)))  float  v8f;
typedef int v4i __attribute__((vector_size(16)));   // matches builtin's int4 param

// merge two online-softmax partial states (policy max/sum + ref-at-argmax, ref max/sum)
__device__ __forceinline__ void merge_stats(
    float& pmax, float& psum, float& refat, float& rmax, float& rsum,
    float bpmax, float bpsum, float brefat, float brmax, float brsum)
{
  float nm = fmaxf(pmax, bpmax);
  psum = psum * __expf(pmax - nm) + bpsum * __expf(bpmax - nm);
  if (bpmax > pmax) refat = brefat;
  pmax = nm;
  float nr = fmaxf(rmax, brmax);
  rsum = rsum * __expf(rmax - nr) + brsum * __expf(brmax - nr);
  rmax = nr;
}

// 16-lane butterfly stats over one 16-column stripe.
__device__ __forceinline__ void stripe_stats(
    float pv, float rv,
    float& pmax, float& psum, float& refat, float& rmax, float& rsum)
{
  pmax = pv; refat = rv;
#pragma unroll
  for (int off = 1; off < 16; off <<= 1) {
    float opm = __shfl_xor(pmax, off, 16);
    float orf = __shfl_xor(refat, off, 16);
    if (opm > pmax) { pmax = opm; refat = orf; }
  }
  psum = __expf(pv - pmax);
#pragma unroll
  for (int off = 1; off < 16; off <<= 1) psum += __shfl_xor(psum, off, 16);
  rmax = rv;
#pragma unroll
  for (int off = 1; off < 16; off <<= 1) rmax = fmaxf(rmax, __shfl_xor(rmax, off, 16));
  rsum = __expf(rv - rmax);
#pragma unroll
  for (int off = 1; off < 16; off <<= 1) rsum += __shfl_xor(rsum, off, 16);
}

// One K-step of WMMA work: 4 row-tiles x 2 stripes x 2 models = 16 WMMAs.
__device__ __forceinline__ void wmma_step(
    const __bf16 (&sA)[2][2][BM][KS], const __bf16 (&sB)[2][2][BN][KS],
    int buf, int wave, int ln16, int hi,
    v8f (&accP)[4][2], v8f (&accR)[4][2])
{
  v16bf fBp[2], fBr[2];
#pragma unroll
  for (int s = 0; s < 2; ++s) {
    int ncol = wave * 32 + s * 16 + ln16;
#pragma unroll
    for (int e = 0; e < 16; ++e) {
      int k = (e & 7) + ((e >= 8) ? 16 : 0) + hi * 8;
      fBp[s][e] = sB[buf][0][ncol][k];
      fBr[s][e] = sB[buf][1][ncol][k];
    }
  }
#pragma unroll
  for (int r = 0; r < 4; ++r) {
    v16bf fAp, fAr;
#pragma unroll
    for (int e = 0; e < 16; ++e) {         // ISA 16-bit A layout (16x32)
      int j = e >> 1, h = e & 1;
      int k = ((j & 3) << 1) + h + ((j >= 4) ? 16 : 0) + (hi ? 8 : 0);
      fAp[e] = sA[buf][0][r * 16 + ln16][k];
      fAr[e] = sA[buf][1][r * 16 + ln16][k];
    }
#pragma unroll
    for (int s = 0; s < 2; ++s) {
      accP[r][s] = __builtin_amdgcn_wmma_f32_16x16x32_bf16(
          false, fAp, false, fBp[s], (short)0, accP[r][s], false, false);
      accR[r][s] = __builtin_amdgcn_wmma_f32_16x16x32_bf16(
          false, fAr, false, fBr[s], (short)0, accR[r][s], false, false);
    }
  }
}

// Per-iteration stats: per-wave butterfly partials -> fold into running state.
__device__ __forceinline__ void stats_phase(
    v8f (&accP)[4][2], v8f (&accR)[4][2],
    float (&wpart)[8][BM][6], float (&runp)[BM][6],
    int tid, int wave, int ln16, int hi, int it)
{
#pragma unroll
  for (int r = 0; r < 4; ++r) {
#pragma unroll
    for (int i = 0; i < 8; ++i) {
      float pmax, psum, refat, rmax, rsum;
      stripe_stats(accP[r][0][i], accR[r][0][i], pmax, psum, refat, rmax, rsum);
      float pmax1, psum1, refat1, rmax1, rsum1;
      stripe_stats(accP[r][1][i], accR[r][1][i], pmax1, psum1, refat1, rmax1, rsum1);
      merge_stats(pmax, psum, refat, rmax, rsum, pmax1, psum1, refat1, rmax1, rsum1);
      if (ln16 == 0) {
        int rloc = r * 16 + i + 8 * hi;
        wpart[wave][rloc][0] = pmax;
        wpart[wave][rloc][1] = psum;
        wpart[wave][rloc][2] = refat;
        wpart[wave][rloc][3] = rmax;
        wpart[wave][rloc][4] = rsum;
      }
    }
  }
  __syncthreads();
  if (tid < BM) {
    float pmax, psum, refat, rmax, rsum;
    if (it == 0) { pmax = NEG_HUGE; psum = 0.f; refat = 0.f; rmax = NEG_HUGE; rsum = 0.f; }
    else {
      pmax = runp[tid][0]; psum = runp[tid][1]; refat = runp[tid][2];
      rmax = runp[tid][3]; rsum = runp[tid][4];
    }
#pragma unroll
    for (int wv = 0; wv < 8; ++wv)
      merge_stats(pmax, psum, refat, rmax, rsum,
                  wpart[wv][tid][0], wpart[wv][tid][1], wpart[wv][tid][2],
                  wpart[wv][tid][3], wpart[wv][tid][4]);
    runp[tid][0] = pmax; runp[tid][1] = psum; runp[tid][2] = refat;
    runp[tid][3] = rmax; runp[tid][4] = rsum;
  }
  __syncthreads();
}

// -----------------------------------------------------------------------------
// Kernel 0: bulk fp32 -> bf16 conversion into workspace.
// -----------------------------------------------------------------------------
__global__ __launch_bounds__(256) void cvt_f32_to_bf16(
    const float* __restrict__ src, __bf16* __restrict__ dst, int n8)
{
  int i = blockIdx.x * blockDim.x + threadIdx.x;
  if (i >= n8) return;
  const float4* p = (const float4*)src + 2 * (size_t)i;
  float4 a = p[0], b = p[1];
  v8bf o;
  o[0] = (__bf16)a.x; o[1] = (__bf16)a.y; o[2] = (__bf16)a.z; o[3] = (__bf16)a.w;
  o[4] = (__bf16)b.x; o[5] = (__bf16)b.y; o[6] = (__bf16)b.z; o[7] = (__bf16)b.w;
  *(v8bf*)(dst + 8 * (size_t)i) = o;
}

// -----------------------------------------------------------------------------
// Kernel 1 (primary): bf16-native fused dual GEMM + streaming log-softmax stats.
// Staging uses GLOBAL_LOAD_ASYNC_TO_LDS_B128 (no VGPR staging, ASYNCcnt
// tracked). Double buffered, one barrier per K-step. launch_bounds(256,1)
// frees the full wave32 register file (no scratch spills).
// -----------------------------------------------------------------------------
__global__ __launch_bounds__(256, 1) void grpo_gemm_stats_bf16(
    const __bf16* __restrict__ xb,  const __bf16* __restrict__ wb,
    const __bf16* __restrict__ rxb, const __bf16* __restrict__ rwb,
    float* __restrict__ partials)
{
  __shared__ __bf16 sA[2][2][BM][KS];   // [buf][model][row][k]   16 KB
  __shared__ __bf16 sB[2][2][BN][KS];   // [buf][model][col][k]   64 KB
  __shared__ float  wpart[8][BM][6];
  __shared__ float  runp[BM][6];

  const int chunk = blockIdx.x;
  const int rb    = blockIdx.y;
  const int tid   = threadIdx.x;
  const int wave  = tid >> 5;
  const int lane  = tid & 31;
  const int ln16  = lane & 15;
  const int hi    = lane >> 4;
  const int row0  = rb * BM;

  const int a_r  = tid >> 2;            // 0..63 row in A tile
  const int a_kg = (tid & 3) * 8;       // 8 contiguous bf16 per thread per model
  // B: thread tid owns vocab row (n0 + tid), all 32 k, per model (4x b128)

  v8f accP[4][2], accR[4][2];
  int n0 = 0;

#if HAVE_ASYNC_LDS
  // Async LDS DMA staging: 10x b128 per thread per K-step, zero staging VGPRs.
  auto async_stage = [&](int k0, int buf) {
    __builtin_amdgcn_global_load_async_to_lds_b128(
        (AS1 v4i*)(xb + (size_t)(row0 + a_r) * H_DIM + k0 + a_kg),
        (AS3 v4i*)&sA[buf][0][a_r][a_kg], 0, 0);
    __builtin_amdgcn_global_load_async_to_lds_b128(
        (AS1 v4i*)(rxb + (size_t)(row0 + a_r) * H_DIM + k0 + a_kg),
        (AS3 v4i*)&sA[buf][1][a_r][a_kg], 0, 0);
#pragma unroll
    for (int u = 0; u < 4; ++u) {
      __builtin_amdgcn_global_load_async_to_lds_b128(
          (AS1 v4i*)(wb + (size_t)(n0 + tid) * H_DIM + k0 + 8 * u),
          (AS3 v4i*)&sB[buf][0][tid][8 * u], 0, 0);
      __builtin_amdgcn_global_load_async_to_lds_b128(
          (AS1 v4i*)(rwb + (size_t)(n0 + tid) * H_DIM + k0 + 8 * u),
          (AS3 v4i*)&sB[buf][1][tid][8 * u], 0, 0);
    }
  };
#else
  uint4 rA[2], rB[2][4];
  auto load_regs = [&](int k0) {
    rA[0] = *(const uint4*)&xb [(size_t)(row0 + a_r) * H_DIM + k0 + a_kg];
    rA[1] = *(const uint4*)&rxb[(size_t)(row0 + a_r) * H_DIM + k0 + a_kg];
    const uint4* pw = (const uint4*)&wb [(size_t)(n0 + tid) * H_DIM + k0];
    const uint4* pq = (const uint4*)&rwb[(size_t)(n0 + tid) * H_DIM + k0];
#pragma unroll
    for (int u = 0; u < 4; ++u) { rB[0][u] = pw[u]; rB[1][u] = pq[u]; }
  };
  auto store_lds = [&](int buf) {
    *(uint4*)&sA[buf][0][a_r][a_kg] = rA[0];
    *(uint4*)&sA[buf][1][a_r][a_kg] = rA[1];
    uint4* dp = (uint4*)&sB[buf][0][tid][0];
    uint4* dq = (uint4*)&sB[buf][1][tid][0];
#pragma unroll
    for (int u = 0; u < 4; ++u) { dp[u] = rB[0][u]; dq[u] = rB[1][u]; }
  };
#endif

  for (int it = 0; it < NITER; ++it) {
    n0 = chunk * VCHUNK + it * BN;
#pragma unroll
    for (int r = 0; r < 4; ++r)
#pragma unroll
      for (int s = 0; s < 2; ++s)
#pragma unroll
        for (int i = 0; i < 8; ++i) { accP[r][s][i] = 0.f; accR[r][s][i] = 0.f; }

#if HAVE_ASYNC_LDS
    async_stage(0, 0);
    __builtin_amdgcn_s_wait_asynccnt(0);
    __syncthreads();
    for (int ks = 0; ks < NK; ++ks) {
      const int cur = ks & 1;
      if (ks + 1 < NK) async_stage((ks + 1) * KS, cur ^ 1);  // DMA next buffer
      wmma_step(sA, sB, cur, wave, ln16, hi, accP, accR);    // ds_load + wmma
      if (ks + 1 < NK) __builtin_amdgcn_s_wait_asynccnt(0);  // DMA landed
      __syncthreads();
    }
#else
    load_regs(0);
    store_lds(0);
    __syncthreads();
    for (int ks = 0; ks < NK; ++ks) {
      const int cur = ks & 1;
      if (ks + 1 < NK) load_regs((ks + 1) * KS);
      wmma_step(sA, sB, cur, wave, ln16, hi, accP, accR);
      if (ks + 1 < NK) store_lds(cur ^ 1);
      __syncthreads();
    }
#endif
    stats_phase(accP, accR, wpart, runp, tid, wave, ln16, hi, it);
  }

  if (tid < BM) {
    float* dst = &partials[((size_t)chunk * M_DIM + row0 + tid) * PSTRIDE];
    dst[0] = runp[tid][0]; dst[1] = runp[tid][1]; dst[2] = runp[tid][2];
    dst[3] = runp[tid][3]; dst[4] = runp[tid][4];
  }
}

// -----------------------------------------------------------------------------
// Kernel 1 (fallback, if workspace too small for bf16 staging): fp32 inputs
// converted on the fly while staging into LDS.
// -----------------------------------------------------------------------------
__global__ __launch_bounds__(256, 1) void grpo_gemm_stats_f32(
    const float* __restrict__ x,   const float* __restrict__ wt,
    const float* __restrict__ rx,  const float* __restrict__ rwt,
    float* __restrict__ partials)
{
  __shared__ __bf16 sA[2][2][BM][KS];
  __shared__ __bf16 sB[2][2][BN][KS];
  __shared__ float  wpart[8][BM][6];
  __shared__ float  runp[BM][6];

  const int chunk = blockIdx.x;
  const int rb    = blockIdx.y;
  const int tid   = threadIdx.x;
  const int wave  = tid >> 5;
  const int lane  = tid & 31;
  const int ln16  = lane & 15;
  const int hi    = lane >> 4;
  const int row0  = rb * BM;
  const int a_r  = tid >> 2;
  const int a_kg = (tid & 3) * 8;

  v8f accP[4][2], accR[4][2];
  float4 rAp[2], rAr[2], rBp[8], rBr[8];
  int n0 = 0;

  auto load_regs = [&](int k0) {
    const float* pa = &x [(size_t)(row0 + a_r) * H_DIM + k0 + a_kg];
    const float* pr = &rx[(size_t)(row0 + a_r) * H_DIM + k0 + a_kg];
    rAp[0] = ((const float4*)pa)[0]; rAp[1] = ((const float4*)pa)[1];
    rAr[0] = ((const float4*)pr)[0]; rAr[1] = ((const float4*)pr)[1];
    const float* pw = &wt [(size_t)(n0 + tid) * H_DIM + k0];
    const float* pq = &rwt[(size_t)(n0 + tid) * H_DIM + k0];
#pragma unroll
    for (int u = 0; u < 8; ++u) { rBp[u] = ((const float4*)pw)[u]; rBr[u] = ((const float4*)pq)[u]; }
  };

  auto store_lds = [&](int buf) {
#pragma unroll
    for (int u = 0; u < 2; ++u) {
      sA[buf][0][a_r][a_kg + 4*u + 0] = (__bf16)rAp[u].x;
      sA[buf][0][a_r][a_kg + 4*u + 1] = (__bf16)rAp[u].y;
      sA[buf][0][a_r][a_kg + 4*u + 2] = (__bf16)rAp[u].z;
      sA[buf][0][a_r][a_kg + 4*u + 3] = (__bf16)rAp[u].w;
      sA[buf][1][a_r][a_kg + 4*u + 0] = (__bf16)rAr[u].x;
      sA[buf][1][a_r][a_kg + 4*u + 1] = (__bf16)rAr[u].y;
      sA[buf][1][a_r][a_kg + 4*u + 2] = (__bf16)rAr[u].z;
      sA[buf][1][a_r][a_kg + 4*u + 3] = (__bf16)rAr[u].w;
    }
#pragma unroll
    for (int u = 0; u < 8; ++u) {
      sB[buf][0][tid][4*u + 0] = (__bf16)rBp[u].x;
      sB[buf][0][tid][4*u + 1] = (__bf16)rBp[u].y;
      sB[buf][0][tid][4*u + 2] = (__bf16)rBp[u].z;
      sB[buf][0][tid][4*u + 3] = (__bf16)rBp[u].w;
      sB[buf][1][tid][4*u + 0] = (__bf16)rBr[u].x;
      sB[buf][1][tid][4*u + 1] = (__bf16)rBr[u].y;
      sB[buf][1][tid][4*u + 2] = (__bf16)rBr[u].z;
      sB[buf][1][tid][4*u + 3] = (__bf16)rBr[u].w;
    }
  };

  for (int it = 0; it < NITER; ++it) {
    n0 = chunk * VCHUNK + it * BN;
#pragma unroll
    for (int r = 0; r < 4; ++r)
#pragma unroll
      for (int s = 0; s < 2; ++s)
#pragma unroll
        for (int i = 0; i < 8; ++i) { accP[r][s][i] = 0.f; accR[r][s][i] = 0.f; }

    load_regs(0);
    store_lds(0);
    __syncthreads();
    for (int ks = 0; ks < NK; ++ks) {
      const int cur = ks & 1;
      if (ks + 1 < NK) load_regs((ks + 1) * KS);
      wmma_step(sA, sB, cur, wave, ln16, hi, accP, accR);
      if (ks + 1 < NK) store_lds(cur ^ 1);
      __syncthreads();
    }
    stats_phase(accP, accR, wpart, runp, tid, wave, ln16, hi, it);
  }

  if (tid < BM) {
    float* dst = &partials[((size_t)chunk * M_DIM + row0 + tid) * PSTRIDE];
    dst[0] = runp[tid][0]; dst[1] = runp[tid][1]; dst[2] = runp[tid][2];
    dst[3] = runp[tid][3]; dst[4] = runp[tid][4];
  }
}

// -----------------------------------------------------------------------------
// Kernel 2: merge the NCHUNK partials per row -> tok_logp and ref_tok_logp.
// -----------------------------------------------------------------------------
__global__ __launch_bounds__(256) void grpo_row_reduce(
    const float* __restrict__ partials, float* __restrict__ rowstats)
{
  int m = blockIdx.x * blockDim.x + threadIdx.x;
  if (m >= M_DIM) return;
  float pmax = NEG_HUGE, psum = 0.f, refat = 0.f, rmax = NEG_HUGE, rsum = 0.f;
  for (int c = 0; c < NCHUNK; ++c) {
    const float* p = &partials[((size_t)c * M_DIM + m) * PSTRIDE];
    merge_stats(pmax, psum, refat, rmax, rsum, p[0], p[1], p[2], p[3], p[4]);
  }
  rowstats[2 * m + 0] = -__logf(psum);
  rowstats[2 * m + 1] = refat - rmax - __logf(rsum);
}

// -----------------------------------------------------------------------------
// Kernel 3: advantages + k3 KL + masked loss and mean_kl (2 scalars out).
// -----------------------------------------------------------------------------
__global__ __launch_bounds__(256) void grpo_finalize(
    const float* __restrict__ rowstats, const int* __restrict__ amask,
    const float* __restrict__ rewards, float* __restrict__ out)
{
  __shared__ float s_adv[B_DIM];
  __shared__ float s_loss, s_mask;
  __shared__ float s_klb[B_DIM], s_mb[B_DIM];
  const int tid = threadIdx.x;
  if (tid == 0) { s_loss = 0.f; s_mask = 0.f; }
  if (tid < B_DIM) { s_klb[tid] = 0.f; s_mb[tid] = 0.f; }
  if (tid == 0) {
    for (int g = 0; g < B_DIM / G_GEN; ++g) {
      float mean = 0.f;
      for (int j = 0; j < G_GEN; ++j) mean += rewards[g * G_GEN + j];
      mean *= (1.f / G_GEN);
      float var = 0.f;
      for (int j = 0; j < G_GEN; ++j) {
        float d = rewards[g * G_GEN + j] - mean;
        var += d * d;
      }
      var /= (G_GEN - 1);               // unbiased std, like torch.std
      float sd = sqrtf(var);
      for (int j = 0; j < G_GEN; ++j)
        s_adv[g * G_GEN + j] = (rewards[g * G_GEN + j] - mean) / (sd + EPS_C);
    }
  }
  __syncthreads();
  float lloss = 0.f, lmask = 0.f;
  for (int m = tid; m < M_DIM; m += blockDim.x) {
    int b = m / T_DIM;
    float mk  = (float)amask[m];
    float tok = rowstats[2 * m + 0];
    float ref = rowstats[2 * m + 1];
    float d   = ref - tok;
    float kl  = __expf(d) - d - 1.f;
    float pt  = s_adv[b] - BETA_C * kl;   // exp(tok - sg(tok)) == 1 in value
    lloss += pt * mk;
    lmask += mk;
    atomicAdd(&s_klb[b], kl * mk);
    atomicAdd(&s_mb[b], mk);
  }
  atomicAdd(&s_loss, lloss);
  atomicAdd(&s_mask, lmask);
  __syncthreads();
  if (tid == 0) {
    float mkl = 0.f;
    for (int b = 0; b < B_DIM; ++b) mkl += s_klb[b] / s_mb[b];
    out[0] = -s_loss / s_mask;
    out[1] = mkl / B_DIM;
  }
}

// -----------------------------------------------------------------------------
extern "C" void kernel_launch(void* const* d_in, const int* in_sizes, int n_in,
                              void* d_out, int out_size, void* d_ws, size_t ws_size,
                              hipStream_t stream) {
  (void)in_sizes; (void)n_in; (void)out_size;
  const float* x    = (const float*)d_in[0];   // (B,T,H)
  const float* w    = (const float*)d_in[1];   // (V,H)
  const int*   am   = (const int*)  d_in[2];   // (B,T)
  const float* rew  = (const float*)d_in[3];   // (B,)
  const float* rx   = (const float*)d_in[4];   // (B,T,H)
  const float* rwt  = (const float*)d_in[5];   // (V,H)
  float* out = (float*)d_out;

  float* ws       = (float*)d_ws;
  float* partials = ws;                                      // NCHUNK*M*PSTRIDE floats (~1.6 MB)
  float* rowstats = ws + (size_t)NCHUNK * M_DIM * PSTRIDE;   // M*2 floats

  const size_t statBytes  = ((size_t)NCHUNK * M_DIM * PSTRIDE + 2 * M_DIM) * sizeof(float);
  const size_t bf16Off    = (statBytes + 255) & ~(size_t)255;
  const size_t nW         = (size_t)V_DIM * H_DIM;           // 65,536,000
  const size_t nX         = (size_t)M_DIM * H_DIM;           //  4,194,304
  const size_t bf16Bytes  = (2 * nW + 2 * nX) * 2;           // ~266 MB
  const bool   preconvert = (ws_size >= bf16Off + bf16Bytes);

  dim3 g1(NCHUNK, NROWB);
  if (preconvert) {
    __bf16* base = (__bf16*)((char*)d_ws + bf16Off);
    __bf16* wb   = base;
    __bf16* rwb  = wb  + nW;
    __bf16* xb   = rwb + nW;
    __bf16* rxb  = xb  + nX;
    const int nW8 = (int)(nW / 8), nX8 = (int)(nX / 8);
    cvt_f32_to_bf16<<<dim3(nW8 / 256), dim3(256), 0, stream>>>(w,   wb,  nW8);
    cvt_f32_to_bf16<<<dim3(nW8 / 256), dim3(256), 0, stream>>>(rwt, rwb, nW8);
    cvt_f32_to_bf16<<<dim3(nX8 / 256), dim3(256), 0, stream>>>(x,   xb,  nX8);
    cvt_f32_to_bf16<<<dim3(nX8 / 256), dim3(256), 0, stream>>>(rx,  rxb, nX8);
    grpo_gemm_stats_bf16<<<g1, dim3(256), 0, stream>>>(xb, wb, rxb, rwb, partials);
  } else {
    grpo_gemm_stats_f32<<<g1, dim3(256), 0, stream>>>(x, w, rx, rwt, partials);
  }
  grpo_row_reduce<<<dim3(M_DIM / 256), dim3(256), 0, stream>>>(partials, rowstats);
  grpo_finalize<<<dim3(1), dim3(256), 0, stream>>>(rowstats, am, rew, out);
}
